// MultiHeadAttention_61907658605242
// MI455X (gfx1250) — compile-verified, matrix operand order checked by reading, not run on hardware
//
#include <hip/hip_runtime.h>
#include <hip/hip_bf16.h>
#include <math.h>

// ---------------------------------------------------------------------------
// MHA forward for MI455X (gfx1250, wave32, WMMA + TDM).
//   B=2, S=2048, D_MODEL=1024, H=16, D_K=64
// ---------------------------------------------------------------------------

typedef _Float16 half_t;
typedef __attribute__((ext_vector_type(16))) _Float16 v16h;
typedef __attribute__((ext_vector_type(8)))  _Float16 v8h;
typedef __attribute__((ext_vector_type(8)))  float    v8f;
typedef __attribute__((ext_vector_type(4)))  unsigned int u32x4;
typedef __attribute__((ext_vector_type(8)))  int i32x8;
typedef __attribute__((ext_vector_type(4)))  int i32x4;

#define MHA_B 2
#define MHA_S 2048
#define MHA_D 1024
#define MHA_H 16
#define MHA_DK 64
#define MHA_M (MHA_B * MHA_S)   // 4096
#define MHA_NEG -1e9f

// WMMA 16x16x32 f16->f32 wrapper (8-arg form, probe-confirmed).
__device__ __forceinline__ v8f wmma_f32_16x16x32_f16(v16h a, v16h b, v8f c) {
    return __builtin_amdgcn_wmma_f32_16x16x32_f16(
        /*neg_a=*/false, a, /*neg_b=*/false, b,
        /*c_mod=*/(short)0, c, /*reuse_a=*/false, /*reuse_b=*/false);
}

// Load a 16x32 WMMA A-fragment (or 32x16 B-fragment, layouts are symmetric)
// from a row-major matrix (ISA 7.12.2):
//   element e<8  : k = col0 + hsel*8 + e
//   element e>=8 : k = col0 + 16 + hsel*8 + (e-8)
template <typename T>
__device__ __forceinline__ v16h load_frag_rowmajor(const T* __restrict__ src,
                                                   int row0, int col0, int ld,
                                                   int lane) {
    const int hsel = lane >> 4;
    const int r = lane & 15;
    const T* p = src + (size_t)(row0 + r) * ld + col0 + (hsel << 3);
    v16h out;
    if constexpr (sizeof(T) == 4) {
        v8f lo = *(const v8f*)p;
        v8f hi = *(const v8f*)(p + 16);
#pragma unroll
        for (int i = 0; i < 8; ++i) {
            out[i]     = (_Float16)lo[i];
            out[i + 8] = (_Float16)hi[i];
        }
    } else {
        v8h lo = *(const v8h*)p;
        v8h hi = *(const v8h*)(p + 16);
#pragma unroll
        for (int i = 0; i < 8; ++i) {
            out[i]     = lo[i];
            out[i + 8] = hi[i];
        }
    }
    return out;
}

// Fragment straight out of LDS (two contiguous 16B groups).
__device__ __forceinline__ v16h frag_from_lds(const _Float16* p) {
    v8h lo = *(const v8h*)p;
    v8h hi = *(const v8h*)(p + 16);
    v16h out;
#pragma unroll
    for (int i = 0; i < 8; ++i) { out[i] = lo[i]; out[i + 8] = hi[i]; }
    return out;
}

// ---------------------------------------------------------------------------
// Tensor Data Mover: 2D tile load Global -> LDS (D# per ISA 8.3/8.4).
//   data_size = 2 bytes; tile (tile_d0 x tile_d1); dim0 stride in elements.
// lds_off is the low-32 generic-pointer bits (== LDS byte offset, ISA 10.2).
// ---------------------------------------------------------------------------
__device__ __forceinline__ void tdm_load_2d_f16(unsigned long long gaddr,
                                                unsigned int lds_off,
                                                int tile_d0, int tile_d1,
                                                long long stride0) {
    u32x4 g0;
    g0[0] = 1u;                                            // count=1 (valid D#)
    g0[1] = lds_off;                                       // lds_addr (bytes)
    g0[2] = (unsigned int)(gaddr & 0xFFFFFFFFull);         // global_addr[31:0]
    g0[3] = (unsigned int)((gaddr >> 32) & 0x1FFFFFFull)   // global_addr[56:32]
            | (2u << 30);                                  // type=2 ("image")
    i32x8 g1;
    g1[0] = (1 << 16);                                     // data_size=1 -> 2B
    g1[1] = (tile_d0 & 0xFFFF) << 16;                      // tensor_dim0[15:0]
    g1[2] = ((unsigned)tile_d0 >> 16) | ((tile_d1 & 0xFFFF) << 16); // dim0 hi | dim1 lo
    g1[3] = ((unsigned)tile_d1 >> 16) | ((tile_d0 & 0xFFFF) << 16); // dim1 hi | tile_dim0
    g1[4] = (tile_d1 & 0xFFFF);                            // tile_dim1; tile_dim2=0
    g1[5] = (int)(stride0 & 0xFFFFFFFFll);                 // tensor_dim0_stride lo
    g1[6] = (int)((stride0 >> 32) & 0xFFFFll);             // stride hi; dim1_stride=0
    g1[7] = 0;
    i32x4 z4 = {0, 0, 0, 0};
#if __clang_major__ >= 23
    i32x8 z8 = {0, 0, 0, 0, 0, 0, 0, 0};
    __builtin_amdgcn_tensor_load_to_lds(g0, g1, z4, z4, z8, 0);
#else
    __builtin_amdgcn_tensor_load_to_lds(g0, g1, z4, z4, 0);
#endif
}

// ---------------------------------------------------------------------------
// GEMM: out[M,N] = X[M,K] @ W[N,K]^T + bias[N]
// One wave computes a 32x64 tile: 2 A-frags x 4 B-frags -> 8 WMMAs / k-step.
// TRANS_OUT stores f16 out[N][M]: accumulator index v maps to consecutive m,
// so each lane emits one packed v8h (16B) store per tile.
// ---------------------------------------------------------------------------
#define GEMM_WAVES 8

template <typename AT, typename OT, bool TRANS_OUT>
__global__ __launch_bounds__(256) void mha_gemm_xwT(
    const AT* __restrict__ X, const float* __restrict__ W,
    const float* __restrict__ bias, OT* __restrict__ out,
    int M, int N, int K) {
    const int lane = threadIdx.x & 31;
    const int wv   = threadIdx.x >> 5;
    const int tilesN = N >> 6;
    const int gw = blockIdx.x * GEMM_WAVES + wv;
    const int tm = (gw / tilesN) << 5;
    const int tn = (gw % tilesN) << 6;
    if (tm >= M) return;

    const int hsel = lane >> 4;
    const int r = lane & 15;

    v8f c[2][4];
#pragma unroll
    for (int i = 0; i < 2; ++i)
#pragma unroll
        for (int j = 0; j < 4; ++j) c[i][j] = (v8f)0.0f;

    for (int k0 = 0; k0 < K; k0 += 32) {
        if (k0 + 32 < K) {  // gfx1250 global_prefetch_b8 for next k-step
            __builtin_prefetch(X + (size_t)(tm + r) * K + k0 + 32, 0, 3);
            __builtin_prefetch(W + (size_t)(tn + r) * K + k0 + 32, 0, 3);
        }
        v16h a0 = load_frag_rowmajor(X, tm, k0, K, lane);
        v16h a1 = load_frag_rowmajor(X, tm + 16, k0, K, lane);
        v16h b0 = load_frag_rowmajor(W, tn, k0, K, lane);
        v16h b1 = load_frag_rowmajor(W, tn + 16, k0, K, lane);
        v16h b2 = load_frag_rowmajor(W, tn + 32, k0, K, lane);
        v16h b3 = load_frag_rowmajor(W, tn + 48, k0, K, lane);
        c[0][0] = wmma_f32_16x16x32_f16(a0, b0, c[0][0]);
        c[0][1] = wmma_f32_16x16x32_f16(a0, b1, c[0][1]);
        c[0][2] = wmma_f32_16x16x32_f16(a0, b2, c[0][2]);
        c[0][3] = wmma_f32_16x16x32_f16(a0, b3, c[0][3]);
        c[1][0] = wmma_f32_16x16x32_f16(a1, b0, c[1][0]);
        c[1][1] = wmma_f32_16x16x32_f16(a1, b1, c[1][1]);
        c[1][2] = wmma_f32_16x16x32_f16(a1, b2, c[1][2]);
        c[1][3] = wmma_f32_16x16x32_f16(a1, b3, c[1][3]);
    }

    // C/D layout: VGPR v -> row (tm + i*16 + hsel*8 + v), lane r -> column.
#pragma unroll
    for (int i = 0; i < 2; ++i) {
#pragma unroll
        for (int j = 0; j < 4; ++j) {
            const int col = tn + (j << 4) + r;
            const float bv = bias[col];
            if constexpr (TRANS_OUT) {
                v8h pack;
#pragma unroll
                for (int v = 0; v < 8; ++v)
                    pack[v] = (_Float16)(c[i][j][v] + bv);
                *(v8h*)(out + (size_t)col * M + tm + (i << 4) + (hsel << 3)) = pack;
            } else {
#pragma unroll
                for (int v = 0; v < 8; ++v) {
                    const int row = tm + (i << 4) + (hsel << 3) + v;
                    const float val = c[i][j][v] + bv;
                    if constexpr (sizeof(OT) == 2)
                        out[(size_t)row * N + col] = (OT)(_Float16)val;
                    else
                        out[(size_t)row * N + col] = (OT)val;
                }
            }
        }
    }
}

// ---------------------------------------------------------------------------
// Flash-style attention. One block (8 waves) owns one (b,h) and 256 queries
// (2 q-tiles per wave). K/V chunks (32 keys) are staged Global->LDS by the
// Tensor Data Mover (wave 0 issues double-buffered tensor_load_to_lds;
// s_wait_tensorcnt drains; workgroup barriers publish). V is pre-transposed
// (Vt[n][m]) so its TDM tile and LDS fragments are contiguous.
// 1/sqrt(dk)=0.125 is folded into the Q fragments (exact in f16).
// ---------------------------------------------------------------------------
#define ATT_WAVES 8
#define ATT_QT 2
#define ATT_CHUNK 32

__global__ __launch_bounds__(256) void mha_attention(
    const half_t* __restrict__ Qp, const half_t* __restrict__ Kp,
    const half_t* __restrict__ Vt, const int* __restrict__ mask,
    half_t* __restrict__ ctx) {
    __shared__ __align__(16) _Float16 kbuf[2][ATT_CHUNK][MHA_DK];   // 8 KB
    __shared__ __align__(16) _Float16 vbuf[2][MHA_DK][ATT_CHUNK];   // 8 KB
    __shared__ __align__(16) _Float16 pbuf[ATT_WAVES][ATT_QT][16][32]; // 16 KB

    const int lane = threadIdx.x & 31;
    const int wv   = threadIdx.x >> 5;
    const int hsel = lane >> 4;
    const int r = lane & 15;

    const int blkPerBH = MHA_S / (16 * ATT_QT * ATT_WAVES);  // 8
    const int qblk = blockIdx.x % blkPerBH;
    const int h = (blockIdx.x / blkPerBH) % MHA_H;
    const int b = blockIdx.x / (blkPerBH * MHA_H);
    const int qrow0 = qblk * (16 * ATT_QT * ATT_WAVES) + wv * (16 * ATT_QT);

    const half_t* Qb  = Qp + (size_t)b * MHA_S * MHA_D + h * MHA_DK;
    const half_t* Kb  = Kp + (size_t)b * MHA_S * MHA_D + h * MHA_DK;
    const half_t* VtB = Vt + (size_t)(h * MHA_DK) * MHA_M + b * MHA_S;

    // Q tiles with 1/sqrt(dk) folded in (power of two -> exact in f16).
    v16h qa[ATT_QT][2];
#pragma unroll
    for (int t = 0; t < ATT_QT; ++t) {
        qa[t][0] = load_frag_rowmajor(Qb, qrow0 + (t << 4), 0, MHA_D, lane);
        qa[t][1] = load_frag_rowmajor(Qb, qrow0 + (t << 4), 32, MHA_D, lane);
#pragma unroll
        for (int i = 0; i < 16; ++i) {
            qa[t][0][i] *= (_Float16)0.125f;
            qa[t][1][i] *= (_Float16)0.125f;
        }
    }

    float mrun[ATT_QT][8], lrun[ATT_QT][8];
    v8f o[ATT_QT][4];
#pragma unroll
    for (int t = 0; t < ATT_QT; ++t) {
#pragma unroll
        for (int v = 0; v < 8; ++v) { mrun[t][v] = -1e30f; lrun[t][v] = 0.0f; }
#pragma unroll
        for (int g = 0; g < 4; ++g) o[t][g] = (v8f)0.0f;
    }

    // TDM: preload chunk 0 into buffer 0 (wave 0 only; EXEC ignored by TDM).
    if (wv == 0) {
        tdm_load_2d_f16((unsigned long long)(uintptr_t)Kb,
                        (unsigned int)(uintptr_t)&kbuf[0][0][0],
                        MHA_DK, ATT_CHUNK, MHA_D);
        tdm_load_2d_f16((unsigned long long)(uintptr_t)VtB,
                        (unsigned int)(uintptr_t)&vbuf[0][0][0],
                        ATT_CHUNK, MHA_DK, MHA_M);
    }

    for (int j0 = 0, it = 0; j0 < MHA_S; j0 += ATT_CHUNK, ++it) {
        const int cur = it & 1;
        const int nxt = cur ^ 1;
        if (wv == 0) {
            // Issue next chunk's pair (dummy re-load of chunk 0 on the last
            // iteration keeps the tensorcnt<=2 invariant; it targets the
            // buffer nobody reads this iteration).
            const int jn = (j0 + ATT_CHUNK < MHA_S) ? j0 + ATT_CHUNK : 0;
            tdm_load_2d_f16((unsigned long long)(uintptr_t)(Kb + (size_t)jn * MHA_D),
                            (unsigned int)(uintptr_t)&kbuf[nxt][0][0],
                            MHA_DK, ATT_CHUNK, MHA_D);
            tdm_load_2d_f16((unsigned long long)(uintptr_t)(VtB + jn),
                            (unsigned int)(uintptr_t)&vbuf[nxt][0][0],
                            ATT_CHUNK, MHA_DK, MHA_M);
            // <=2 outstanding == current chunk's pair has landed.
            __builtin_amdgcn_s_wait_tensorcnt(2);
        }
        __syncthreads();   // publish kbuf[cur]/vbuf[cur] to all waves

        // Score B-fragments from LDS: lane r = key row, contiguous d.
        v16h kb0a = frag_from_lds(&kbuf[cur][r][(hsel << 3)]);
        v16h kb0b = frag_from_lds(&kbuf[cur][r][32 + (hsel << 3)]);
        v16h kb1a = frag_from_lds(&kbuf[cur][16 + r][(hsel << 3)]);
        v16h kb1b = frag_from_lds(&kbuf[cur][16 + r][32 + (hsel << 3)]);

#pragma unroll
        for (int t = 0; t < ATT_QT; ++t) {
            v8f s0 = (v8f)0.0f, s1 = (v8f)0.0f;
            s0 = wmma_f32_16x16x32_f16(qa[t][0], kb0a, s0);
            s0 = wmma_f32_16x16x32_f16(qa[t][1], kb0b, s0);
            s1 = wmma_f32_16x16x32_f16(qa[t][0], kb1a, s1);
            s1 = wmma_f32_16x16x32_f16(qa[t][1], kb1b, s1);

            // Mask + online softmax. Element v is row
            // (qrow0 + t*16 + hsel*8 + v); s0 col = j0+r, s1 col = j0+16+r.
#pragma unroll
            for (int v = 0; v < 8; ++v) {
                const int mrow = qrow0 + (t << 4) + (hsel << 3) + v;
                float sv0 = s0[v];
                float sv1 = s1[v];
                if (mask[(size_t)mrow * MHA_S + j0 + r] == 0)      sv0 = MHA_NEG;
                if (mask[(size_t)mrow * MHA_S + j0 + 16 + r] == 0) sv1 = MHA_NEG;

                // Row reductions: xor 1/2/4/8 stays inside each wave32 half.
                float mx = fmaxf(sv0, sv1);
#pragma unroll
                for (int off = 8; off >= 1; off >>= 1)
                    mx = fmaxf(mx, __shfl_xor(mx, off, 32));

                const float mnew = fmaxf(mrun[t][v], mx);
                const float corr = __expf(mrun[t][v] - mnew);
                const float p0 = __expf(sv0 - mnew);
                const float p1 = __expf(sv1 - mnew);

                float rs = p0 + p1;
#pragma unroll
                for (int off = 8; off >= 1; off >>= 1)
                    rs += __shfl_xor(rs, off, 32);

                lrun[t][v] = lrun[t][v] * corr + rs;
                mrun[t][v] = mnew;
#pragma unroll
                for (int g = 0; g < 4; ++g) o[t][g][v] *= corr;

                // Stage P (C layout) into LDS row-major for A-frag reload.
                pbuf[wv][t][(hsel << 3) + v][r]      = (_Float16)p0;
                pbuf[wv][t][(hsel << 3) + v][16 + r] = (_Float16)p1;
            }
        }

        // Reload P as 16x32 A-fragments (per-wave LDS, DS ops in-order).
        v16h pa[ATT_QT];
#pragma unroll
        for (int t = 0; t < ATT_QT; ++t) {
            const _Float16* prow = &pbuf[wv][t][r][0];
            pa[t] = frag_from_lds(prow + (hsel << 3));
            v8h hi = *(const v8h*)(prow + 16 + (hsel << 3));
#pragma unroll
            for (int i = 0; i < 8; ++i) pa[t][i + 8] = hi[i];
        }

        // O[q][d] += P[q][kk] * V[kk][d]: vbuf rows are d, cols are keys.
#pragma unroll
        for (int g = 0; g < 4; ++g) {
            v16h vb = frag_from_lds(&vbuf[cur][(g << 4) + r][(hsel << 3)]);
#pragma unroll
            for (int t = 0; t < ATT_QT; ++t)
                o[t][g] = wmma_f32_16x16x32_f16(pa[t], vb, o[t][g]);
        }

        __syncthreads();   // all waves done with buf[cur] before TDM reuses it
    }

    // Normalize and store ctx (f16) in [B,S,D] layout.
    half_t* Cb = ctx + (size_t)b * MHA_S * MHA_D + h * MHA_DK;
#pragma unroll
    for (int t = 0; t < ATT_QT; ++t) {
#pragma unroll
        for (int v = 0; v < 8; ++v) {
            const float inv = 1.0f / lrun[t][v];
            const int row = qrow0 + (t << 4) + (hsel << 3) + v;
#pragma unroll
            for (int g = 0; g < 4; ++g)
                Cb[(size_t)row * MHA_D + (g << 4) + r] = (half_t)(o[t][g][v] * inv);
        }
    }
}

// ---------------------------------------------------------------------------
// Launch: 3 projection GEMMs (V transposed) -> attention -> output GEMM.
// Inputs: q,k,v,mask,Wq,bq,Wk,bk,Wv,bv,Wo,bo
// ---------------------------------------------------------------------------
extern "C" void kernel_launch(void* const* d_in, const int* in_sizes, int n_in,
                              void* d_out, int out_size, void* d_ws, size_t ws_size,
                              hipStream_t stream) {
    (void)in_sizes; (void)n_in; (void)out_size; (void)ws_size;

    const float* q    = (const float*)d_in[0];
    const float* k    = (const float*)d_in[1];
    const float* v    = (const float*)d_in[2];
    const int*   mask = (const int*)d_in[3];
    const float* Wq   = (const float*)d_in[4];
    const float* bq   = (const float*)d_in[5];
    const float* Wk   = (const float*)d_in[6];
    const float* bk   = (const float*)d_in[7];
    const float* Wv   = (const float*)d_in[8];
    const float* bv   = (const float*)d_in[9];
    const float* Wo   = (const float*)d_in[10];
    const float* bo   = (const float*)d_in[11];
    float* out = (float*)d_out;

    const size_t elems = (size_t)MHA_M * MHA_D;  // 4,194,304
    half_t* Qp  = (half_t*)d_ws;
    half_t* Kp  = Qp + elems;
    half_t* Vt  = Kp + elems;                    // transposed: [D][M]
    half_t* ctx = Vt + elems;

    const int M = MHA_M;  // 4096
    const int N = MHA_D;  // 1024
    const int K = MHA_D;  // 1024
    const int gemmBlocks = (M / 32) * (N / 64) / GEMM_WAVES;          // 256
    const int attBlocks  = MHA_B * MHA_H *
                           (MHA_S / (16 * ATT_QT * ATT_WAVES));       // 256

    mha_gemm_xwT<float, half_t, false><<<gemmBlocks, 256, 0, stream>>>(q, Wq, bq, Qp, M, N, K);
    mha_gemm_xwT<float, half_t, false><<<gemmBlocks, 256, 0, stream>>>(k, Wk, bk, Kp, M, N, K);
    mha_gemm_xwT<float, half_t, true ><<<gemmBlocks, 256, 0, stream>>>(v, Wv, bv, Vt, M, N, K);

    mha_attention<<<attBlocks, 256, 0, stream>>>(Qp, Kp, Vt, mask, ctx);

    mha_gemm_xwT<half_t, float, false><<<gemmBlocks, 256, 0, stream>>>(ctx, Wo, bo, out, M, N, K);
}
